// EncoderLayer_7988639170612
// MI455X (gfx1250) — matrix operand order repaired and checked by reading, not compiled, not run on hardware
//
#include <hip/hip_runtime.h>

typedef _Float16 half_t;
typedef __attribute__((ext_vector_type(16))) _Float16 v16h;
typedef __attribute__((ext_vector_type(8)))  _Float16 v8h;
typedef __attribute__((ext_vector_type(8)))  float    v8f;
typedef unsigned int u32x4 __attribute__((ext_vector_type(4)));
typedef unsigned int u32x8 __attribute__((ext_vector_type(8)));

// ---------------------------------------------------------------------------
// WMMA helpers (CDNA5 16x16x32 f16 -> f32)
// ---------------------------------------------------------------------------
__device__ __forceinline__ v8f wmma16x16x32(v16h a, v16h b, v8f c) {
  return __builtin_amdgcn_wmma_f32_16x16x32_f16(
      /*neg_a=*/false, a, /*neg_b=*/false, b,
      /*c_mod=*/(short)0, c, /*reuse_a=*/false, /*reuse_b=*/false);
}

// A fragment, this code's contiguous k order (B uses the same): hb = lane>>4, row = lane&15.
// half index i: K = i + hb*16. One contiguous 32B LDS read.
__device__ __forceinline__ v16h load_a_frag(const half_t* base, int row, int ldk,
                                            int kbase, int hb) {
  return *reinterpret_cast<const v16h*>(base + row * ldk + kbase + hb * 16);
}

// B fragment, same contiguous k order as A: col = lane&15, b[i] = B[K = i + 16*hb][col].
// LDS holds B K-major per output column -> one contiguous 32B read.
__device__ __forceinline__ v16h load_bt_frag(const half_t* baseT, int col, int ldk,
                                             int kbase, int hb) {
  return *reinterpret_cast<const v16h*>(baseT + col * ldk + kbase + hb * 16);
}

// ---------------------------------------------------------------------------
// CDNA5 async/TDM data movement
// ---------------------------------------------------------------------------
__device__ __forceinline__ unsigned lds_offset_of(const void* p) {
  return (unsigned)(unsigned long long)p;  // LDS offset lives in addr[31:0]
}

// GLOBAL_LOAD_ASYNC_TO_LDS_B128 (GV form): per-lane 16B global -> LDS.
__device__ __forceinline__ void async_load_b128(unsigned lds_off, const void* gaddr) {
  asm volatile("global_load_async_to_lds_b128 %0, %1, off"
               :: "v"(lds_off), "v"(gaddr) : "memory");
}
__device__ __forceinline__ void wait_asynccnt0() {
  asm volatile("s_wait_asynccnt 0x0" ::: "memory");
}

// TENSOR_LOAD_TO_LDS of a 2D f16 tile: tileR x tileC elements from a row-major
// tensor [rows x cols] starting at gtile, into contiguous LDS (row-major,
// leading dim = tileC). D# groups per ISA 8.3/8.4 (2-group form, 2D tensor).
__device__ __forceinline__ void tdm_load_tile_f16(unsigned lds_off,
                                                  const half_t* gtile,
                                                  unsigned cols_total,
                                                  unsigned rows_total,
                                                  unsigned tileC, unsigned tileR) {
  const unsigned long long ga = (unsigned long long)gtile;
  u32x4 g0;
  g0[0] = 1u;                                    // count=1, user descriptor
  g0[1] = lds_off;                               // lds_addr
  g0[2] = (unsigned)ga;                          // global_addr[31:0]
  g0[3] = ((unsigned)(ga >> 32) & 0x01FFFFFFu)   // global_addr[56:32]
          | 0x80000000u;                         // type=2 ("image")
  u32x8 g1;
  g1[0] = 0x00010000u;                           // wg_mask=0, data_size=1 (2B)
  g1[1] = (cols_total & 0xFFFFu) << 16;          // tensor_dim0[15:0]
  g1[2] = ((cols_total >> 16) & 0xFFFFu)         // tensor_dim0[31:16]
          | ((rows_total & 0xFFFFu) << 16);      // tensor_dim1[15:0]
  g1[3] = ((rows_total >> 16) & 0xFFFFu)         // tensor_dim1[31:16]
          | (tileC << 16);                       // tile_dim0
  g1[4] = tileR;                                 // tile_dim1 (tile_dim2=0)
  g1[5] = cols_total;                            // tensor_dim0_stride[31:0]
  g1[6] = 0u;                                    // stride0[47:32], stride1[15:0]
  g1[7] = 0u;
  asm volatile("tensor_load_to_lds %0, %1" :: "s"(g0), "s"(g1) : "memory");
}

// ---------------------------------------------------------------------------
// f32 -> f16 conversion (plain and transposing)
// ---------------------------------------------------------------------------
__global__ __launch_bounds__(256) void cvt_f32_f16_kernel(
    const float* __restrict__ in, half_t* __restrict__ out, int n) {
  int i = blockIdx.x * 256 + threadIdx.x;
  if (i < n) out[i] = (half_t)in[i];
}

// in: [K][N] f32 row-major  ->  out: [N][K] f16 row-major
__global__ __launch_bounds__(256) void cvt_transpose_kernel(
    const float* __restrict__ in, half_t* __restrict__ out, int K, int N) {
  __shared__ float tile[32][33];
  const int kb = blockIdx.x * 32, nb = blockIdx.y * 32;
  const int tx = threadIdx.x & 31, ty = threadIdx.x >> 5;
#pragma unroll
  for (int i = ty; i < 32; i += 8)
    tile[i][tx] = in[(size_t)(kb + i) * N + nb + tx];
  __syncthreads();
#pragma unroll
  for (int i = ty; i < 32; i += 8)
    out[(size_t)(nb + i) * K + kb + tx] = (half_t)tile[tx][i];
}

// ---------------------------------------------------------------------------
// GEMM: C[M,N] = A[M,K] * Bt[N,K]^T   (A f16 row-major, Bt = B pre-transposed)
// epi 0: store f16 C ; epi 1: f32 (C+bias+resid) ; epi 2: f16 relu(C+bias)
// Block: 256 thr (8 waves), tile 256x64, K-step 64, double-buffered LDS.
// Each wave computes 32x64 (8 accumulators, 16 WMMA / stage; B-frag reused
// across two row groups). A tile via async-to-LDS (one row per thread),
// B tile via one TDM tensor_load per stage; transfers overlap compute.
// ---------------------------------------------------------------------------
__global__ __launch_bounds__(256) void gemm_wmma_kernel(
    const half_t* __restrict__ A, const half_t* __restrict__ Bt,
    float* __restrict__ Cf, half_t* __restrict__ Ch,
    const float* __restrict__ bias, const float* __restrict__ resid,
    int M, int N, int K, int epi) {
  __shared__ __align__(64) half_t As[2][256 * 64];  // 2 x 32 KB
  __shared__ __align__(64) half_t Bs[2][64 * 64];   // 2 x  8 KB

  const int tileM = blockIdx.y * 256;
  const int tileN = blockIdx.x * 64;
  const int t    = threadIdx.x;
  const int wave = t >> 5;
  const int lane = t & 31;
  const int hb   = lane >> 4;
  const int lr   = lane & 15;

  const unsigned asOff0 = lds_offset_of(As[0]);
  const unsigned asOff1 = lds_offset_of(As[1]);
  const unsigned bsOff0 = lds_offset_of(Bs[0]);
  const unsigned bsOff1 = lds_offset_of(Bs[1]);

  auto issue_stage = [&](int k0, int buf) {
    if (wave == 0) {
      tdm_load_tile_f16(buf ? bsOff1 : bsOff0, Bt + (size_t)tileN * K + k0,
                        (unsigned)K, (unsigned)N, 64u, 64u);
    }
    // A tile 256x64: thread t owns row t (8 x async b128).
    const half_t* gA = A + (size_t)(tileM + t) * K + k0;
    const unsigned dst = (buf ? asOff1 : asOff0) + (unsigned)t * 128u;
#pragma unroll
    for (int c = 0; c < 8; ++c)
      async_load_b128(dst + (unsigned)c * 16u, gA + c * 8);
  };

  const v8f zero8 = {0.f, 0.f, 0.f, 0.f, 0.f, 0.f, 0.f, 0.f};
  v8f acc[2][4];
#pragma unroll
  for (int rg = 0; rg < 2; ++rg)
#pragma unroll
    for (int j = 0; j < 4; ++j) acc[rg][j] = zero8;

  const int S = K / 64;
  issue_stage(0, 0);

  for (int s = 0; s < S; ++s) {
    // Drain this stage's transfers, then one rendezvous.
    wait_asynccnt0();
    if (wave == 0) __builtin_amdgcn_s_wait_tensorcnt(0);
    __syncthreads();
    // Kick off next stage into the other buffer; runs under the WMMAs below.
    if (s + 1 < S) issue_stage((s + 1) * 64, (s + 1) & 1);

    const half_t* Ac = As[s & 1];
    const half_t* Bc = Bs[s & 1];
#pragma unroll
    for (int ks = 0; ks < 2; ++ks) {
      const v16h a0 = load_a_frag(Ac, wave * 32 + lr, 64, ks * 32, hb);
      const v16h a1 = load_a_frag(Ac, wave * 32 + 16 + lr, 64, ks * 32, hb);
#pragma unroll
      for (int j = 0; j < 4; ++j) {
        const v16h bf = load_bt_frag(Bc, j * 16 + lr, 64, ks * 32, hb);
        acc[0][j] = wmma16x16x32(a0, bf, acc[0][j]);
        acc[1][j] = wmma16x16x32(a1, bf, acc[1][j]);
      }
    }
  }

  // Epilogue. C/D layout: row M = v + 8*hb, col N = lr (per 16x16 subtile j).
#pragma unroll
  for (int rg = 0; rg < 2; ++rg) {
#pragma unroll
    for (int j = 0; j < 4; ++j) {
#pragma unroll
      for (int v = 0; v < 8; ++v) {
        const int gm = tileM + wave * 32 + rg * 16 + v + hb * 8;
        const int gn = tileN + j * 16 + lr;
        const size_t idx = (size_t)gm * N + gn;
        float val = acc[rg][j][v];
        if (epi == 0) {
          Ch[idx] = (half_t)val;
        } else if (epi == 1) {
          Cf[idx] = val + bias[gn] + resid[idx];
        } else {
          float r = val + bias[gn];
          Ch[idx] = (half_t)(r > 0.f ? r : 0.f);
        }
      }
    }
  }
}

// ---------------------------------------------------------------------------
// Flash attention: one block per (b, h, 64-query tile). 4 waves x 16 rows.
// Streaming softmax over 64-key chunks; S and O never leave VGPR/LDS.
// ---------------------------------------------------------------------------
__global__ __launch_bounds__(128) void attn_kernel(
    const half_t* __restrict__ Q, const half_t* __restrict__ Kmat,
    const half_t* __restrict__ V, const float* __restrict__ mask,
    half_t* __restrict__ O, int N, int D) {
  constexpr int Dh = 64;
  __shared__ __align__(64) half_t Qs[64 * 64];
  __shared__ __align__(64) half_t Ks[64 * 64];
  __shared__ __align__(64) half_t Vt[64 * 64];    // transposed: Vt[d][key]
  __shared__ __align__(64) half_t Ps[4][16 * 64]; // per-wave P staging

  const int qt = blockIdx.x, h = blockIdx.y, b = blockIdx.z;
  const int t = threadIdx.x, wave = t >> 5, lane = t & 31;
  const int hb = lane >> 4, lr = lane & 15;
  const int q0 = qt * 64;
  const size_t baseBH = ((size_t)b * N) * D + (size_t)h * Dh;
  const float scale = 0.125f;  // 1/sqrt(64)

  {  // Q tile (64x64)
    const int r = t >> 1, seg = (t & 1) * 32;
    const half_t* src = Q + baseBH + (size_t)(q0 + r) * D + seg;
    *reinterpret_cast<v16h*>(&Qs[r * 64 + seg])      = *reinterpret_cast<const v16h*>(src);
    *reinterpret_cast<v16h*>(&Qs[r * 64 + seg + 16]) = *reinterpret_cast<const v16h*>(src + 16);
  }

  const v8f zero8 = {0.f, 0.f, 0.f, 0.f, 0.f, 0.f, 0.f, 0.f};
  v8f oacc[4];
#pragma unroll
  for (int j = 0; j < 4; ++j) oacc[j] = zero8;
  float mrow[8], lrow[8];
#pragma unroll
  for (int v = 0; v < 8; ++v) { mrow[v] = -3.0e38f; lrow[v] = 0.f; }

  half_t* pw = &Ps[wave][0];

  for (int kt = 0; kt < N; kt += 64) {
    __syncthreads();
    {  // K chunk, row-major [key][d]
      const int r = t >> 1, seg = (t & 1) * 32;
      const half_t* src = Kmat + baseBH + (size_t)(kt + r) * D + seg;
      *reinterpret_cast<v16h*>(&Ks[r * 64 + seg])      = *reinterpret_cast<const v16h*>(src);
      *reinterpret_cast<v16h*>(&Ks[r * 64 + seg + 16]) = *reinterpret_cast<const v16h*>(src + 16);
    }
    {  // V chunk, coalesced read, transposed LDS write
      const int key = t & 63, dg = (t >> 6) * 32;
      const half_t* src = V + baseBH + (size_t)(kt + key) * D + dg;
      const v16h a0 = *reinterpret_cast<const v16h*>(src);
      const v16h a1 = *reinterpret_cast<const v16h*>(src + 16);
#pragma unroll
      for (int i = 0; i < 16; ++i) Vt[(dg + i) * 64 + key] = a0[i];
#pragma unroll
      for (int i = 0; i < 16; ++i) Vt[(dg + 16 + i) * 64 + key] = a1[i];
    }
    __syncthreads();

    // S = Q * K^T for this wave's 16 rows x 64 keys
    v8f s[4];
#pragma unroll
    for (int j = 0; j < 4; ++j) s[j] = zero8;
#pragma unroll
    for (int ks = 0; ks < 2; ++ks) {
      const v16h a = load_a_frag(Qs + wave * 16 * 64, lr, 64, ks * 32, hb);
#pragma unroll
      for (int j = 0; j < 4; ++j) {
        const v16h bf = load_bt_frag(Ks, j * 16 + lr, 64, ks * 32, hb);
        s[j] = wmma16x16x32(a, bf, s[j]);
      }
    }

    // scale + mask; per-row chunk max
    float sv[4][8];
    float cmax[8];
#pragma unroll
    for (int v = 0; v < 8; ++v) cmax[v] = -3.0e38f;
#pragma unroll
    for (int j = 0; j < 4; ++j) {
#pragma unroll
      for (int v = 0; v < 8; ++v) {
        const int qr = q0 + wave * 16 + v + hb * 8;
        const int kc = kt + j * 16 + lr;
        const float val = s[j][v] * scale + mask[((size_t)b * N + qr) * N + kc];
        sv[j][v] = val;
        cmax[v] = fmaxf(cmax[v], val);
      }
    }
    // 16-lane row reductions (rows live per half-wave, cols across 16 lanes)
#pragma unroll
    for (int v = 0; v < 8; ++v) {
      float m = cmax[v];
      m = fmaxf(m, __shfl_xor(m, 1, 32));
      m = fmaxf(m, __shfl_xor(m, 2, 32));
      m = fmaxf(m, __shfl_xor(m, 4, 32));
      m = fmaxf(m, __shfl_xor(m, 8, 32));
      const float mnew  = fmaxf(mrow[v], m);
      const float alpha = __expf(mrow[v] - mnew);
      mrow[v] = mnew;
      float su = 0.f;
#pragma unroll
      for (int j = 0; j < 4; ++j) {
        const float p = __expf(sv[j][v] - mnew);
        sv[j][v] = p;
        su += p;
        oacc[j][v] *= alpha;
      }
      su += __shfl_xor(su, 1, 32);
      su += __shfl_xor(su, 2, 32);
      su += __shfl_xor(su, 4, 32);
      su += __shfl_xor(su, 8, 32);
      lrow[v] = lrow[v] * alpha + su;
    }

    // Stage P (f16) in this wave's private LDS tile (same-wave DS ops in order)
#pragma unroll
    for (int j = 0; j < 4; ++j)
#pragma unroll
      for (int v = 0; v < 8; ++v)
        pw[(v + hb * 8) * 64 + j * 16 + lr] = (half_t)sv[j][v];

    // O += P * V
#pragma unroll
    for (int ks = 0; ks < 2; ++ks) {
      const v16h a = load_a_frag(pw, lr, 64, ks * 32, hb);
#pragma unroll
      for (int j = 0; j < 4; ++j) {
        const v16h bf = load_bt_frag(Vt, j * 16 + lr, 64, ks * 32, hb);
        oacc[j] = wmma16x16x32(a, bf, oacc[j]);
      }
    }
  }

  // Normalize and store f16 in [B,N,D] layout (head-interleaved)
#pragma unroll
  for (int j = 0; j < 4; ++j) {
#pragma unroll
    for (int v = 0; v < 8; ++v) {
      const int qr = q0 + wave * 16 + v + hb * 8;
      const int dc = j * 16 + lr;
      O[baseBH + (size_t)qr * D + dc] = (half_t)(oacc[j][v] / lrow[v]);
    }
  }
}

// ---------------------------------------------------------------------------
// LayerNorm over last dim (D=1024), one block per row.
// ---------------------------------------------------------------------------
__global__ __launch_bounds__(256) void layernorm_kernel(
    const float* __restrict__ in, const float* __restrict__ g,
    const float* __restrict__ be, float* __restrict__ outf,
    half_t* __restrict__ outh, int D) {
  __shared__ float red0[8], red1[8];
  const int row = blockIdx.x;
  const float* x = in + (size_t)row * D;
  const int t = threadIdx.x;

  float s = 0.f, s2 = 0.f;
  for (int i = t; i < D; i += 256) { const float v = x[i]; s += v; s2 += v * v; }
#pragma unroll
  for (int m = 1; m < 32; m <<= 1) { s += __shfl_xor(s, m, 32); s2 += __shfl_xor(s2, m, 32); }
  if ((t & 31) == 0) { red0[t >> 5] = s; red1[t >> 5] = s2; }
  __syncthreads();
  float ts = 0.f, ts2 = 0.f;
#pragma unroll
  for (int i = 0; i < 8; ++i) { ts += red0[i]; ts2 += red1[i]; }
  const float mean = ts / D;
  const float var  = ts2 / D - mean * mean;
  const float inv  = rsqrtf(var + 1e-5f);

  for (int i = t; i < D; i += 256) {
    const float y = (x[i] - mean) * inv * g[i] + be[i];
    if (outf) outf[(size_t)row * D + i] = y;
    if (outh) outh[(size_t)row * D + i] = (half_t)y;
  }
}

// ---------------------------------------------------------------------------
// Host orchestration
// ---------------------------------------------------------------------------
extern "C" void kernel_launch(void* const* d_in, const int* in_sizes, int n_in,
                              void* d_out, int out_size, void* d_ws, size_t ws_size,
                              hipStream_t stream) {
  const int B = 4, N = 2048, D = 1024, H = 16, F = 2048;
  const int M = B * N;  // 8192 rows

  const float* x    = (const float*)d_in[0];
  const float* mask = (const float*)d_in[1];
  const float* Wq   = (const float*)d_in[2];
  const float* Wk   = (const float*)d_in[3];
  const float* Wv   = (const float*)d_in[4];
  const float* Wc   = (const float*)d_in[5];
  const float* bc   = (const float*)d_in[6];
  const float* W1   = (const float*)d_in[7];
  const float* b1   = (const float*)d_in[8];
  const float* W2   = (const float*)d_in[9];
  const float* b2   = (const float*)d_in[10];
  const float* g1   = (const float*)d_in[11];
  const float* be1  = (const float*)d_in[12];
  const float* g2   = (const float*)d_in[13];
  const float* be2  = (const float*)d_in[14];

  char* ws = (char*)d_ws;
  size_t off = 0;
  auto alloc = [&](size_t bytes) -> void* {
    off = (off + 255) & ~(size_t)255;
    void* p = ws + off;
    off += bytes;
    return p;
  };

  half_t* xh    = (half_t*)alloc((size_t)M * D * 2);
  half_t* Wqt   = (half_t*)alloc((size_t)D * D * 2);  // [N][K] transposed weights
  half_t* Wkt   = (half_t*)alloc((size_t)D * D * 2);
  half_t* Wvt   = (half_t*)alloc((size_t)D * D * 2);
  half_t* Wct   = (half_t*)alloc((size_t)D * D * 2);
  half_t* W1t   = (half_t*)alloc((size_t)D * F * 2);  // [F][D]
  half_t* W2t   = (half_t*)alloc((size_t)F * D * 2);  // [D][F]
  half_t* Qh    = (half_t*)alloc((size_t)M * D * 2);
  half_t* Kh    = (half_t*)alloc((size_t)M * D * 2);
  half_t* Vh    = (half_t*)alloc((size_t)M * D * 2);
  half_t* attnh = (half_t*)alloc((size_t)M * D * 2);
  float*  res1  = (float*)alloc((size_t)M * D * 4);
  float*  x1f   = (float*)alloc((size_t)M * D * 4);
  // Aliased reuse (stream-ordered, producers finish before consumers):
  half_t* x1h  = attnh;  // LN1 f16 out reuses attn buffer (free after Wc GEMM)
  half_t* midh = Qh;     // FFN mid (M*F halves) reuses Qh+Kh (free after attention)
  float*  res2 = res1;   // second residual reuses first (free after LN1)

  cvt_f32_f16_kernel<<<(M * D + 255) / 256, 256, 0, stream>>>(x, xh, M * D);
  auto cvtT = [&](const float* s, half_t* d, int K, int Ncols) {
    dim3 g(K / 32, Ncols / 32);
    cvt_transpose_kernel<<<g, 256, 0, stream>>>(s, d, K, Ncols);
  };
  cvtT(Wq, Wqt, D, D);
  cvtT(Wk, Wkt, D, D);
  cvtT(Wv, Wvt, D, D);
  cvtT(Wc, Wct, D, D);
  cvtT(W1, W1t, D, F);
  cvtT(W2, W2t, F, D);

  // Q/K/V projections: [M,D] = [M,D] x [D,D], f16 out
  dim3 gP(D / 64, M / 256);
  gemm_wmma_kernel<<<gP, 256, 0, stream>>>(xh, Wqt, nullptr, Qh, nullptr, nullptr, M, D, D, 0);
  gemm_wmma_kernel<<<gP, 256, 0, stream>>>(xh, Wkt, nullptr, Kh, nullptr, nullptr, M, D, D, 0);
  gemm_wmma_kernel<<<gP, 256, 0, stream>>>(xh, Wvt, nullptr, Vh, nullptr, nullptr, M, D, D, 0);

  // Flash attention
  dim3 gA(N / 64, H, B);
  attn_kernel<<<gA, 128, 0, stream>>>(Qh, Kh, Vh, mask, attnh, N, D);

  // attn @ Wc + bc + x -> res1 (f32)
  gemm_wmma_kernel<<<gP, 256, 0, stream>>>(attnh, Wct, res1, nullptr, bc, x, M, D, D, 1);

  // LN1 -> x1f (f32) + x1h (f16)
  layernorm_kernel<<<M, 256, 0, stream>>>(res1, g1, be1, x1f, x1h, D);

  // FFN up: relu(x1 @ W1 + b1) -> midh (f16), N=F
  dim3 gU(F / 64, M / 256);
  gemm_wmma_kernel<<<gU, 256, 0, stream>>>(x1h, W1t, nullptr, midh, b1, nullptr, M, F, D, 2);

  // FFN down: mid @ W2 + b2 + x1 -> res2 (f32), K=F
  gemm_wmma_kernel<<<gP, 256, 0, stream>>>(midh, W2t, res2, nullptr, b2, x1f, M, D, F, 1);

  // LN2 -> final output (f32)
  layernorm_kernel<<<M, 256, 0, stream>>>(res2, g2, be2, (float*)d_out, nullptr, D);
}
